// Encoder_58600533787127
// MI455X (gfx1250) — matrix operand order repaired and checked by reading, not compiled, not run on hardware
//
#include <hip/hip_runtime.h>

typedef __attribute__((ext_vector_type(16))) _Float16 v16h;
typedef __attribute__((ext_vector_type(8)))  _Float16 v8h;
typedef __attribute__((ext_vector_type(8)))  float    v8f;

#define B_      64
#define NNODES  64
#define NEDGES  4032           // 64*63
#define HID     128

// ---------------------------------------------------------------------------
// Reorder a fp32 weight matrix W[K][N] into WMMA B-fragment order (f16).
// Fragment layout (ISA 7.12.2, 16-bit B 32x16): lane l holds n = l&15,
// k = (l>>4)*8 + e for e < 8, k = 16 + (l>>4)*8 + (e-8) for e >= 8.
// dst[((nt*KC + kc)*32 + lane)*16 + e] = W[kc*32 + k(lane,e)][nt*16 + (lane&15)]
// ---------------------------------------------------------------------------
__global__ void frag_weights(const float* __restrict__ W, _Float16* __restrict__ dst,
                             int K, int N) {
    int idx = blockIdx.x * blockDim.x + threadIdx.x;
    if (idx >= K * N) return;
    int e    = idx & 15;
    int lane = (idx >> 4) & 31;
    int t    = idx >> 9;                 // nt*KC + kc
    int KC   = K >> 5;
    int kc   = t % KC;
    int nt   = t / KC;
    int k    = kc * 32 + ((lane >> 4) << 3) + (e & 7) + ((e >> 3) << 4);
    int n    = nt * 16 + (lane & 15);
    dst[idx] = (_Float16)W[(long)k * N + n];
}

__global__ void cvt_f32_f16(const float* __restrict__ in, _Float16* __restrict__ out, int n) {
    int i = blockIdx.x * blockDim.x + threadIdx.x;
    if (i < n) out[i] = (_Float16)in[i];
}

// ---------------------------------------------------------------------------
// edge2node: m1[b,n,:] = (1/(63+1e-6)) * sum_{s != n} e[b, s*63 + (n<s?n:n-1), :]
// ---------------------------------------------------------------------------
__global__ void edge2node_k(const _Float16* __restrict__ e, _Float16* __restrict__ m1) {
    int bn = blockIdx.x;                 // 0..4095
    int b  = bn >> 6, n = bn & 63;
    int t  = threadIdx.x;                // 0..127 feature
    const _Float16* base = e + (long)b * NEDGES * HID;
    float acc = 0.f;
    #pragma unroll 4
    for (int s = 0; s < NNODES; s++) {
        if (s == n) continue;
        int j = (n < s) ? n : n - 1;
        acc += (float)base[(long)(s * 63 + j) * HID + t];
    }
    m1[(long)bn * HID + t] = (_Float16)(acc * (1.0f / (63.0f + 1e-6f)));
}

// ---------------------------------------------------------------------------
// Fused 2-layer MLP (+ optional third projection to 16 f32 outputs).
// Each wave computes a 16-row tile. Stage1: K = KC0*32 -> 128 (+bias, ReLU),
// staged as f16 in LDS. Stage2: 128 -> 128 (+bias). THIRD: stage2 result back
// to LDS, then 128 -> 16 f32 (+bias) to outF.
// MODE 0: plain rows from inA (stride KC0*32)
// MODE 1: rows = concat(h[sender], h[receiver])                (K0 = 256)
// MODE 2: rows = concat(n[sender], n[receiver], e_skip[row])   (K0 = 384)
// ---------------------------------------------------------------------------
template<int KC0, int MODE, bool THIRD>
__global__ __launch_bounds__(256)
void fused_mlp(const _Float16* __restrict__ inA,
               const _Float16* __restrict__ inE,
               const _Float16* __restrict__ wA, const float* __restrict__ bA,
               const _Float16* __restrict__ wB, const float* __restrict__ bB,
               const _Float16* __restrict__ wC, const float* __restrict__ bC,
               _Float16* __restrict__ outH, float* __restrict__ outF) {
    __shared__ _Float16 lds[8][16 * HID];          // 4KB per wave, 32KB total
    const int lane = threadIdx.x & 31;
    const int wave = threadIdx.x >> 5;
    const int rowtile = blockIdx.x * 8 + wave;
    const int m = lane & 15;                       // A row within tile / C col
    const int g = lane >> 4;                       // K-half group
    const int gr = rowtile * 16 + m;               // this lane's global A row

    const _Float16* seg0;
    const _Float16* seg1;
    const _Float16* seg2;
    if (MODE == 0) {
        seg0 = inA + (long)gr * (KC0 * 32);
        seg1 = seg0; seg2 = seg0;
    } else {
        int b = gr / NEDGES;
        int ei = gr - b * NEDGES;
        int s = ei / 63;
        int j = ei - s * 63;
        int r = j + (j >= s ? 1 : 0);
        seg0 = inA + (long)(b * NNODES + s) * HID;
        seg1 = inA + (long)(b * NNODES + r) * HID;
        seg2 = (MODE == 2) ? (inE + (long)gr * HID) : seg1;
    }

    v8f zero;
    #pragma unroll
    for (int i = 0; i < 8; i++) zero[i] = 0.0f;

    // ---------------- stage 1: K = KC0*32 -> 128, ReLU ----------------
    v8f acc[8];
    #pragma unroll
    for (int nt = 0; nt < 8; nt++) acc[nt] = zero;

    #pragma unroll
    for (int kc = 0; kc < KC0; kc++) {
        const _Float16* ap;
        if (MODE == 0) {
            ap = seg0 + kc * 32 + g * 8;
        } else {
            int seg = (kc * 32) / HID;
            int kin = kc * 32 - seg * HID;
            const _Float16* sp = (seg == 0) ? seg0 : ((seg == 1) ? seg1 : seg2);
            ap = sp + kin + g * 8;
        }
        v8h lo = *(const v8h*)(ap);
        v8h hi = *(const v8h*)(ap + 16);
        v16h a;
        #pragma unroll
        for (int i = 0; i < 8; i++) { a[i] = lo[i]; a[8 + i] = hi[i]; }
        #pragma unroll
        for (int nt = 0; nt < 8; nt++) {
            const _Float16* bp = wA + ((long)((nt * KC0 + kc) * 32 + lane) << 4);
            v8h blo = *(const v8h*)(bp);
            v8h bhi = *(const v8h*)(bp + 8);
            v16h bf;
            #pragma unroll
            for (int i = 0; i < 8; i++) { bf[i] = blo[i]; bf[8 + i] = bhi[i]; }
            acc[nt] = __builtin_amdgcn_wmma_f32_16x16x32_f16(
                false, a, false, bf, (short)0, acc[nt], false, false);
        }
    }

    _Float16* ltile = lds[wave];
    #pragma unroll
    for (int nt = 0; nt < 8; nt++) {
        float bv = bA[nt * 16 + m];
        #pragma unroll
        for (int r = 0; r < 8; r++) {
            float v = acc[nt][r] + bv;
            v = v > 0.f ? v : 0.f;                            // ReLU
            ltile[(r + 8 * g) * HID + nt * 16 + m] = (_Float16)v;
        }
    }

    // ---------------- stage 2: 128 -> 128 (no ReLU) ----------------
    v8f acc2[8];
    #pragma unroll
    for (int nt = 0; nt < 8; nt++) acc2[nt] = zero;

    #pragma unroll
    for (int kc = 0; kc < 4; kc++) {
        const _Float16* ap = ltile + m * HID + kc * 32 + g * 8;
        v8h lo = *(const v8h*)(ap);
        v8h hi = *(const v8h*)(ap + 16);
        v16h a;
        #pragma unroll
        for (int i = 0; i < 8; i++) { a[i] = lo[i]; a[8 + i] = hi[i]; }
        #pragma unroll
        for (int nt = 0; nt < 8; nt++) {
            const _Float16* bp = wB + ((long)((nt * 4 + kc) * 32 + lane) << 4);
            v8h blo = *(const v8h*)(bp);
            v8h bhi = *(const v8h*)(bp + 8);
            v16h bf;
            #pragma unroll
            for (int i = 0; i < 8; i++) { bf[i] = blo[i]; bf[8 + i] = bhi[i]; }
            acc2[nt] = __builtin_amdgcn_wmma_f32_16x16x32_f16(
                false, a, false, bf, (short)0, acc2[nt], false, false);
        }
    }

    if (!THIRD) {
        #pragma unroll
        for (int nt = 0; nt < 8; nt++) {
            float bv = bB[nt * 16 + m];
            #pragma unroll
            for (int r = 0; r < 8; r++) {
                float v = acc2[nt][r] + bv;
                outH[(long)(rowtile * 16 + r + 8 * g) * HID + nt * 16 + m] = (_Float16)v;
            }
        }
    } else {
        // stage2 result -> LDS (overwrite; per-wave DS ops are in-order)
        #pragma unroll
        for (int nt = 0; nt < 8; nt++) {
            float bv = bB[nt * 16 + m];
            #pragma unroll
            for (int r = 0; r < 8; r++) {
                float v = acc2[nt][r] + bv;
                ltile[(r + 8 * g) * HID + nt * 16 + m] = (_Float16)v;
            }
        }
        // ---------------- stage 3: 128 -> 16, f32 out ----------------
        v8f acc3 = zero;
        #pragma unroll
        for (int kc = 0; kc < 4; kc++) {
            const _Float16* ap = ltile + m * HID + kc * 32 + g * 8;
            v8h lo = *(const v8h*)(ap);
            v8h hi = *(const v8h*)(ap + 16);
            v16h a;
            #pragma unroll
            for (int i = 0; i < 8; i++) { a[i] = lo[i]; a[8 + i] = hi[i]; }
            const _Float16* bp = wC + ((long)(kc * 32 + lane) << 4);
            v8h blo = *(const v8h*)(bp);
            v8h bhi = *(const v8h*)(bp + 8);
            v16h bf;
            #pragma unroll
            for (int i = 0; i < 8; i++) { bf[i] = blo[i]; bf[8 + i] = bhi[i]; }
            acc3 = __builtin_amdgcn_wmma_f32_16x16x32_f16(
                false, a, false, bf, (short)0, acc3, false, false);
        }
        float bv = bC[m];
        #pragma unroll
        for (int r = 0; r < 8; r++) {
            outF[(long)(rowtile * 16 + r + 8 * g) * 16 + m] = acc3[r] + bv;
        }
    }
}

// ---------------------------------------------------------------------------
extern "C" void kernel_launch(void* const* d_in, const int* in_sizes, int n_in,
                              void* d_out, int out_size, void* d_ws, size_t ws_size,
                              hipStream_t stream) {
    const float* x    = (const float*)d_in[0];
    // d_in[1]=rel_rec, d_in[2]=rel_send: structure is closed-form, unused
    const float* w1a  = (const float*)d_in[3];  const float* b1a = (const float*)d_in[4];
    const float* w1b  = (const float*)d_in[5];  const float* b1b = (const float*)d_in[6];
    const float* w2a  = (const float*)d_in[7];  const float* b2a = (const float*)d_in[8];
    const float* w2b  = (const float*)d_in[9];  const float* b2b = (const float*)d_in[10];
    const float* w3a  = (const float*)d_in[11]; const float* b3a = (const float*)d_in[12];
    const float* w3b  = (const float*)d_in[13]; const float* b3b = (const float*)d_in[14];
    const float* w4a  = (const float*)d_in[15]; const float* b4a = (const float*)d_in[16];
    const float* w4b  = (const float*)d_in[17]; const float* b4b = (const float*)d_in[18];
    const float* wout = (const float*)d_in[19]; const float* bout= (const float*)d_in[20];
    float* out = (float*)d_out;

    // ---- workspace carve-up (halves, 256B aligned) ----
    char* ws = (char*)d_ws;
    size_t off = 0;
    auto alloc = [&](size_t halves) -> _Float16* {
        _Float16* p = (_Float16*)(ws + off);
        off = (off + halves * 2 + 255) & ~(size_t)255;
        return p;
    };
    _Float16* fw1a = alloc(64  * 128);
    _Float16* fw1b = alloc(128 * 128);
    _Float16* fw2a = alloc(256 * 128);
    _Float16* fw2b = alloc(128 * 128);
    _Float16* fw3a = alloc(128 * 128);
    _Float16* fw3b = alloc(128 * 128);
    _Float16* fw4a = alloc(384 * 128);
    _Float16* fw4b = alloc(128 * 128);
    _Float16* fwo  = alloc(128 * 16);
    _Float16* xh   = alloc((size_t)B_ * NNODES * 64);       // x in f16
    _Float16* hbuf = alloc((size_t)B_ * NNODES * HID);      // node emb h
    _Float16* ebuf = alloc((size_t)B_ * NEDGES * HID);      // edge emb e (66 MB)
    _Float16* m1   = alloc((size_t)B_ * NNODES * HID);      // edge2node
    _Float16* nbuf = alloc((size_t)B_ * NNODES * HID);      // node emb n
    (void)ws_size; (void)in_sizes; (void)n_in; (void)out_size;

    // ---- weight fragmentization + input convert ----
    auto frag = [&](const float* W, _Float16* dst, int K, int N) {
        int tot = K * N;
        frag_weights<<<(tot + 255) / 256, 256, 0, stream>>>(W, dst, K, N);
    };
    frag(w1a, fw1a,  64, 128);  frag(w1b, fw1b, 128, 128);
    frag(w2a, fw2a, 256, 128);  frag(w2b, fw2b, 128, 128);
    frag(w3a, fw3a, 128, 128);  frag(w3b, fw3b, 128, 128);
    frag(w4a, fw4a, 384, 128);  frag(w4b, fw4b, 128, 128);
    frag(wout, fwo, 128, 16);
    {
        int n = B_ * NNODES * 64;
        cvt_f32_f16<<<(n + 255) / 256, 256, 0, stream>>>(x, xh, n);
    }

    // ---- MLP1: x -> h   (4096 rows, K0=64) ----
    fused_mlp<2, 0, false><<<32, 256, 0, stream>>>(
        xh, nullptr, fw1a, b1a, fw1b, b1b, nullptr, nullptr, hbuf, nullptr);

    // ---- MLP2: concat(h[s],h[r]) -> e   (258048 rows, K0=256) ----
    fused_mlp<8, 1, false><<<2016, 256, 0, stream>>>(
        hbuf, nullptr, fw2a, b2a, fw2b, b2b, nullptr, nullptr, ebuf, nullptr);

    // ---- edge2node: e -> m1 ----
    edge2node_k<<<B_ * NNODES, HID, 0, stream>>>(ebuf, m1);

    // ---- MLP3: m1 -> n   (4096 rows, K0=128) ----
    fused_mlp<4, 0, false><<<32, 256, 0, stream>>>(
        m1, nullptr, fw3a, b3a, fw3b, b3b, nullptr, nullptr, nbuf, nullptr);

    // ---- MLP4 + out: concat(n[s],n[r],e) -> 128 -> 128 -> 16 f32 ----
    fused_mlp<12, 2, true><<<2016, 256, 0, stream>>>(
        nbuf, ebuf, fw4a, b4a, fw4b, b4b, fwo, bout, nullptr, out);
}
